// UnfusedMoEExperts_89189290868941
// MI455X (gfx1250) — matrix operand order repaired and checked by reading, not compiled, not run on hardware
//
#include <hip/hip_runtime.h>
#include <hip/hip_bf16.h>
#include <math.h>

// Problem constants (match reference)
constexpr int E = 8, D = 2048, I = 4096, TOPK = 2, N = 4096;
constexpr int CAP = N * TOPK;        // 8192 max entries per expert list
constexpr int BT = 128;              // tile rows (token positions)
constexpr int BC = 128;              // tile cols (output channels)
constexpr int BK = 32;               // K step per WMMA

typedef __attribute__((ext_vector_type(16))) __bf16 v16bf;
typedef __attribute__((ext_vector_type(8)))  float  v8f;
typedef __attribute__((ext_vector_type(4)))  unsigned int vu4;
typedef __attribute__((ext_vector_type(8)))  int vi8;
typedef __attribute__((ext_vector_type(4)))  int vi4;

union Frag16 { v16bf v; uint4 q[2]; };

__device__ __forceinline__ unsigned short f2bf(float f) {
  unsigned int u = __float_as_uint(f);
  unsigned int r = u + 0x7FFFu + ((u >> 16) & 1u);   // round-to-nearest-even
  return (unsigned short)(r >> 16);
}

__device__ __forceinline__ unsigned long long pk4bf(float a, float b, float c, float d) {
  return (unsigned long long)f2bf(a)
       | ((unsigned long long)f2bf(b) << 16)
       | ((unsigned long long)f2bf(c) << 32)
       | ((unsigned long long)f2bf(d) << 48);
}

// SiLU with hardware reciprocal (v_rcp_f32) instead of the IEEE div sequence.
__device__ __forceinline__ float silu_fast(float g) {
  return g * __builtin_amdgcn_rcpf(1.0f + __expf(-g));
}

// Low 32 bits of a generic shared pointer == LDS byte offset (aperture rule).
__device__ __forceinline__ unsigned int ldsOff(const void* p) {
  return (unsigned int)(unsigned long long)p;
}

// ---------------------------------------------------------------------------
// TDM: issue a 2D tile load (rows x 32 bf16 elems, row stride = stride_elems)
// into LDS at lds_off.  D# per CDNA5 ISA 8.3/8.4:
//   group0: count=1 | lds_addr | global_addr | type=2
//   group1: data_size=1(2B), tensor_dim0/1 (no clipping), tile 32 x rows,
//           tensor_dim0_stride (48b, element units)
// ---------------------------------------------------------------------------
__device__ __forceinline__ void tdm_load_tile(unsigned int lds_off,
                                              const unsigned short* gsrc,
                                              unsigned int rows,
                                              unsigned int stride_elems) {
  unsigned long long ga = (unsigned long long)(const void*)gsrc;
  vu4 g0;
  g0.x = 1u;                                   // count=1 (valid user descriptor)
  g0.y = lds_off;                              // LDS byte address
  g0.z = (unsigned int)ga;                     // global addr [31:0]
  g0.w = (unsigned int)(ga >> 32) | (2u << 30);// addr [56:32] | type=2 ("image")
  const unsigned int dim0 = stride_elems;      // row length (>= tile_dim0)
  const unsigned int dim1 = 1u << 20;          // large: no OOB clipping
  vi8 g1;
  g1[0] = (int)0x00010000u;                    // wg_mask=0, data_size=1 (2 bytes)
  g1[1] = (int)((dim0 & 0xFFFFu) << 16);       // tensor_dim0[15:0] (bits 63:48)
  g1[2] = (int)((dim0 >> 16) | ((dim1 & 0xFFFFu) << 16)); // dim0 hi | dim1 lo
  g1[3] = (int)((dim1 >> 16) | ((unsigned int)BK << 16)); // dim1 hi | tile_dim0=32
  g1[4] = (int)(rows & 0xFFFFu);               // tile_dim1 = rows, tile_dim2 = 0
  g1[5] = (int)stride_elems;                   // tensor_dim0_stride [31:0]
  g1[6] = 0;                                   // stride hi / dim1_stride lo
  g1[7] = 0;
  vi4 gz; gz.x = 0; gz.y = 0; gz.z = 0; gz.w = 0;
#if defined(__clang_major__) && (__clang_major__ >= 23)
  vi8 z8; for (int q = 0; q < 8; ++q) z8[q] = 0;
  __builtin_amdgcn_tensor_load_to_lds(g0, g1, gz, gz, z8, 0);
#else
  __builtin_amdgcn_tensor_load_to_lds(g0, g1, gz, gz, 0);
#endif
}

// ---------------------------------------------------------------------------
// Kernel 1: routing. One block, 8 waves; wave e builds expert e's list
// (slot = n*TOPK + k) in ascending slot order -> deterministic, no atomics.
// ---------------------------------------------------------------------------
__global__ void moe_route(const int* __restrict__ eidx, const float* __restrict__ ewt,
                          int* __restrict__ counts, int* __restrict__ lists,
                          float* __restrict__ wlists) {
  const int lane = threadIdx.x & 31;
  const int e    = threadIdx.x >> 5;
  int base = 0;
  for (int it = lane; it < N * TOPK; it += 32) {
    const bool match = (eidx[it] == e);
    const unsigned int m32 = (unsigned int)__ballot(match);
    const int prefix = __popc(m32 & ((1u << lane) - 1u));
    if (match) {
      lists[e * CAP + base + prefix]  = it;
      wlists[e * CAP + base + prefix] = ewt[it];
    }
    base += __popc(m32);
  }
  if (lane == 0) counts[e] = base;
}

// Kernel 2: fp32 -> bf16 bulk convert (used for x always, weights when ws fits)
__global__ void moe_cvt_bf16(const float* __restrict__ src, unsigned short* __restrict__ dst) {
  const size_t i = ((size_t)blockIdx.x * blockDim.x + threadIdx.x) * 4;
  float4 f = *(const float4*)(src + i);
  *(unsigned long long*)(dst + i) = pk4bf(f.x, f.y, f.z, f.w);
}

// ---------------------------------------------------------------------------
// Kernel 3: gate+up GEMM with SwiGLU epilogue.  Double-buffered LDS pipeline;
// B tiles via TDM (bf16 weights) or manual fp32->bf16 staging (fallback).
// ---------------------------------------------------------------------------
template <bool USE_TDM>
__global__ __launch_bounds__(256) void moe_gemm_gu_t(
    const unsigned short* __restrict__ xb,          // [N][D] bf16
    const float* __restrict__ wgf, const float* __restrict__ wuf,   // fp32 (fallback)
    const unsigned short* __restrict__ wgb, const unsigned short* __restrict__ wub, // bf16 (TDM)
    const int* __restrict__ counts, const int* __restrict__ lists,
    unsigned short* __restrict__ h) {               // [CAP][I] bf16
  const int e   = blockIdx.z;
  const int cnt = counts[e];
  const int p0  = blockIdx.x * BT;
  if (p0 >= cnt) return;
  const int i0  = blockIdx.y * BC;

  __shared__ __align__(16) unsigned short As[2][BT * BK];
  __shared__ __align__(16) unsigned short Bg[2][BC * BK];
  __shared__ __align__(16) unsigned short Bu[2][BC * BK];

  const int t    = threadIdx.x;
  const int lane = t & 31;
  const int w    = t >> 5;
  const int wr   = w >> 1;         // wave row tile: wr*32
  const int wc   = w & 1;          // wave col tile: wc*64
  const int ln   = lane & 15;
  const int lh   = lane >> 4;

  const int lrow = t >> 1;         // LDS row this thread stages
  const int lhh  = (t & 1) * 16;   // 16-half K segment

  const unsigned short* aSrc = nullptr;
  {
    const int pos = p0 + lrow;
    if (pos < cnt) {
      const int item = lists[e * CAP + pos];
      aSrc = xb + (size_t)(item >> 1) * D;
    }
  }
  const float* gRow = nullptr;
  const float* uRow = nullptr;
  if constexpr (!USE_TDM) {
    gRow = wgf + ((size_t)e * I + (i0 + lrow)) * D;
    uRow = wuf + ((size_t)e * I + (i0 + lrow)) * D;
  }

  auto stageA = [&](int bi, int k0) {
    uint4 av;
    if (aSrc) av = *(const uint4*)(aSrc + k0 + lhh);
    else { av.x = av.y = av.z = av.w = 0u; }
    *(uint4*)(&As[bi][lrow * BK + lhh]) = av;
  };
  auto stageB = [&](int bi, int k0) {
    if constexpr (USE_TDM) {
      if (w == 0) {      // one wave issues the DMA; EXEC-z waves branch around
        tdm_load_tile(ldsOff(&Bg[bi][0]), wgb + ((size_t)e * I + i0) * D + k0, BC, D);
        tdm_load_tile(ldsOff(&Bu[bi][0]), wub + ((size_t)e * I + i0) * D + k0, BC, D);
      }
    } else {
      const float4* g4 = (const float4*)(gRow + k0 + lhh);
      const float4* u4 = (const float4*)(uRow + k0 + lhh);
      float4 g0 = g4[0], g1 = g4[1], g2 = g4[2], g3 = g4[3];
      float4 u0 = u4[0], u1 = u4[1], u2 = u4[2], u3 = u4[3];
      unsigned long long* dg = (unsigned long long*)(&Bg[bi][lrow * BK + lhh]);
      unsigned long long* du = (unsigned long long*)(&Bu[bi][lrow * BK + lhh]);
      dg[0] = pk4bf(g0.x, g0.y, g0.z, g0.w); dg[1] = pk4bf(g1.x, g1.y, g1.z, g1.w);
      dg[2] = pk4bf(g2.x, g2.y, g2.z, g2.w); dg[3] = pk4bf(g3.x, g3.y, g3.z, g3.w);
      du[0] = pk4bf(u0.x, u0.y, u0.z, u0.w); du[1] = pk4bf(u1.x, u1.y, u1.z, u1.w);
      du[2] = pk4bf(u2.x, u2.y, u2.z, u2.w); du[3] = pk4bf(u3.x, u3.y, u3.z, u3.w);
      if (k0 + BK < D) { __builtin_prefetch(gRow + k0 + BK, 0, 1);
                         __builtin_prefetch(uRow + k0 + BK, 0, 1); }
    }
  };
  auto waitB = [&]() {
    if constexpr (USE_TDM) {
      if (w == 0) __builtin_amdgcn_s_wait_tensorcnt(0);
    }
  };

  v8f accG[2][4], accU[2][4];
  for (int a = 0; a < 2; ++a)
    for (int b = 0; b < 4; ++b)
      for (int q = 0; q < 8; ++q) { accG[a][b][q] = 0.f; accU[a][b][q] = 0.f; }

  stageA(0, 0); stageB(0, 0);
  waitB();
  __syncthreads();

  for (int k0 = 0; k0 < D; k0 += BK) {
    const int buf = (k0 / BK) & 1;
    const bool more = (k0 + BK) < D;
    if (more) { stageA(buf ^ 1, k0 + BK); stageB(buf ^ 1, k0 + BK); }

#pragma unroll
    for (int mt = 0; mt < 2; ++mt) {
      const int arow = wr * 32 + mt * 16 + ln;
      Frag16 fa;
      fa.q[0] = *(const uint4*)(&As[buf][arow * BK + lh * 16]);
      fa.q[1] = *(const uint4*)(&As[buf][arow * BK + lh * 16 + 8]);
#pragma unroll
      for (int nt = 0; nt < 4; ++nt) {
        const int brow = wc * 64 + nt * 16 + ln;
        Frag16 fg, fu;
        const uint4* bg = (const uint4*)(&Bg[buf][brow * BK + lh * 16]);
        const uint4* bu = (const uint4*)(&Bu[buf][brow * BK + lh * 16]);
        fg.q[0] = bg[0]; fg.q[1] = bg[1];
        fu.q[0] = bu[0]; fu.q[1] = bu[1];
        accG[mt][nt] = __builtin_amdgcn_wmma_f32_16x16x32_bf16(
            false, fa.v, false, fg.v, (short)0, accG[mt][nt], false, false);
        accU[mt][nt] = __builtin_amdgcn_wmma_f32_16x16x32_bf16(
            false, fa.v, false, fu.v, (short)0, accU[mt][nt], false, false);
      }
    }
    if (more) { waitB(); __syncthreads(); }
  }

  // epilogue: h = silu(gate) * up, scattered by slot row
#pragma unroll
  for (int mt = 0; mt < 2; ++mt) {
#pragma unroll
    for (int r = 0; r < 8; ++r) {
      const int pos = p0 + wr * 32 + mt * 16 + 8 * lh + r;
      if (pos >= cnt) continue;
      const int item = lists[e * CAP + pos];
      unsigned short* hrow = h + (size_t)item * I;
#pragma unroll
      for (int nt = 0; nt < 4; ++nt) {
        const float g = accG[mt][nt][r];
        const float u = accU[mt][nt][r];
        hrow[i0 + wc * 64 + nt * 16 + ln] = f2bf(silu_fast(g) * u);
      }
    }
  }
}

// ---------------------------------------------------------------------------
// Kernel 4: down-proj GEMM; epilogue scales by routing weight.
// ---------------------------------------------------------------------------
template <bool USE_TDM>
__global__ __launch_bounds__(256) void moe_gemm_down_t(
    const unsigned short* __restrict__ h,           // [CAP][I] bf16
    const float* __restrict__ wdf,                  // fp32 (fallback)
    const unsigned short* __restrict__ wdb,         // bf16 (TDM)
    const int* __restrict__ counts, const int* __restrict__ lists,
    const float* __restrict__ wlists,
    float* __restrict__ contrib) {                  // [CAP][D] f32
  const int e   = blockIdx.z;
  const int cnt = counts[e];
  const int p0  = blockIdx.x * BT;
  if (p0 >= cnt) return;
  const int d0  = blockIdx.y * BC;

  __shared__ __align__(16) unsigned short As[2][BT * BK];
  __shared__ __align__(16) unsigned short Bd[2][BC * BK];

  const int t    = threadIdx.x;
  const int lane = t & 31;
  const int w    = t >> 5;
  const int wr   = w >> 1;
  const int wc   = w & 1;
  const int ln   = lane & 15;
  const int lh   = lane >> 4;

  const int lrow = t >> 1;
  const int lhh  = (t & 1) * 16;

  const unsigned short* aSrc = nullptr;
  {
    const int pos = p0 + lrow;
    if (pos < cnt) {
      const int item = lists[e * CAP + pos];
      aSrc = h + (size_t)item * I;
    }
  }
  const float* dRow = nullptr;
  if constexpr (!USE_TDM) dRow = wdf + ((size_t)e * D + (d0 + lrow)) * I;

  auto stageA = [&](int bi, int k0) {
    uint4 av;
    if (aSrc) av = *(const uint4*)(aSrc + k0 + lhh);
    else { av.x = av.y = av.z = av.w = 0u; }
    *(uint4*)(&As[bi][lrow * BK + lhh]) = av;
  };
  auto stageB = [&](int bi, int k0) {
    if constexpr (USE_TDM) {
      if (w == 0)
        tdm_load_tile(ldsOff(&Bd[bi][0]), wdb + ((size_t)e * D + d0) * I + k0, BC, I);
    } else {
      const float4* d4 = (const float4*)(dRow + k0 + lhh);
      float4 f0 = d4[0], f1 = d4[1], f2 = d4[2], f3 = d4[3];
      unsigned long long* db = (unsigned long long*)(&Bd[bi][lrow * BK + lhh]);
      db[0] = pk4bf(f0.x, f0.y, f0.z, f0.w); db[1] = pk4bf(f1.x, f1.y, f1.z, f1.w);
      db[2] = pk4bf(f2.x, f2.y, f2.z, f2.w); db[3] = pk4bf(f3.x, f3.y, f3.z, f3.w);
      if (k0 + BK < I) __builtin_prefetch(dRow + k0 + BK, 0, 1);
    }
  };
  auto waitB = [&]() {
    if constexpr (USE_TDM) {
      if (w == 0) __builtin_amdgcn_s_wait_tensorcnt(0);
    }
  };

  v8f acc[2][4];
  for (int a = 0; a < 2; ++a)
    for (int b = 0; b < 4; ++b)
      for (int q = 0; q < 8; ++q) acc[a][b][q] = 0.f;

  stageA(0, 0); stageB(0, 0);
  waitB();
  __syncthreads();

  for (int k0 = 0; k0 < I; k0 += BK) {
    const int buf = (k0 / BK) & 1;
    const bool more = (k0 + BK) < I;
    if (more) { stageA(buf ^ 1, k0 + BK); stageB(buf ^ 1, k0 + BK); }

#pragma unroll
    for (int mt = 0; mt < 2; ++mt) {
      const int arow = wr * 32 + mt * 16 + ln;
      Frag16 fa;
      fa.q[0] = *(const uint4*)(&As[buf][arow * BK + lh * 16]);
      fa.q[1] = *(const uint4*)(&As[buf][arow * BK + lh * 16 + 8]);
#pragma unroll
      for (int nt = 0; nt < 4; ++nt) {
        const int brow = wc * 64 + nt * 16 + ln;
        Frag16 fb;
        const uint4* bp = (const uint4*)(&Bd[buf][brow * BK + lh * 16]);
        fb.q[0] = bp[0]; fb.q[1] = bp[1];
        acc[mt][nt] = __builtin_amdgcn_wmma_f32_16x16x32_bf16(
            false, fa.v, false, fb.v, (short)0, acc[mt][nt], false, false);
      }
    }
    if (more) { waitB(); __syncthreads(); }
  }

#pragma unroll
  for (int mt = 0; mt < 2; ++mt) {
#pragma unroll
    for (int r = 0; r < 8; ++r) {
      const int pos = p0 + wr * 32 + mt * 16 + 8 * lh + r;
      if (pos >= cnt) continue;
      const int item = lists[e * CAP + pos];
      const float rw = wlists[e * CAP + pos];
      float* crow = contrib + (size_t)item * D;
#pragma unroll
      for (int nt = 0; nt < 4; ++nt)
        crow[d0 + wc * 64 + nt * 16 + ln] = rw * acc[mt][nt][r];
    }
  }
}

// Kernel 5: out[n] = contrib[2n] + contrib[2n+1]  (fixed order -> deterministic)
__global__ void moe_reduce(const float* __restrict__ contrib, float* __restrict__ out) {
  const int i = (blockIdx.x * blockDim.x + threadIdx.x) * 4;
  const int n = i / D;
  const int d = i % D;
  float4 a = *(const float4*)(contrib + (size_t)(2 * n) * D + d);
  float4 b = *(const float4*)(contrib + (size_t)(2 * n + 1) * D + d);
  float4 o; o.x = a.x + b.x; o.y = a.y + b.y; o.z = a.z + b.z; o.w = a.w + b.w;
  *(float4*)(out + i) = o;
}

// ---------------------------------------------------------------------------
extern "C" void kernel_launch(void* const* d_in, const int* in_sizes, int n_in,
                              void* d_out, int out_size, void* d_ws, size_t ws_size,
                              hipStream_t stream) {
  const float* x    = (const float*)d_in[0];
  const float* ewt  = (const float*)d_in[1];
  const int*   eidx = (const int*)d_in[2];
  // d_in[3] = top_k scalar (TOPK hardcoded)
  const float* wg   = (const float*)d_in[4];
  const float* wu   = (const float*)d_in[5];
  const float* wd   = (const float*)d_in[6];
  float* out        = (float*)d_out;

  char* ws = (char*)d_ws;
  size_t off = 0;
  int*   counts  = (int*)(ws + off);                off += 256;
  int*   lists   = (int*)(ws + off);                off += (size_t)E * CAP * 4;
  float* wlists  = (float*)(ws + off);              off += (size_t)E * CAP * 4;
  unsigned short* xb = (unsigned short*)(ws + off); off += (size_t)N * D * 2;
  unsigned short* h  = (unsigned short*)(ws + off); off += (size_t)CAP * I * 2;
  float* contrib = (float*)(ws + off);              off += (size_t)CAP * D * 4;
  // Optional bf16 weight mirrors for the TDM path (+402 MB)
  const size_t wElems = (size_t)E * I * D;          // per weight tensor
  unsigned short* wgb = (unsigned short*)(ws + off); off += wElems * 2;
  unsigned short* wub = (unsigned short*)(ws + off); off += wElems * 2;
  unsigned short* wdb = (unsigned short*)(ws + off); off += wElems * 2;
  const bool use_tdm = (ws_size >= off);            // fixed per run -> deterministic

  moe_route<<<1, 256, 0, stream>>>(eidx, ewt, counts, lists, wlists);
  moe_cvt_bf16<<<(N * D) / 1024, 256, 0, stream>>>(x, xb);

  dim3 g1(CAP / BT, I / BC, E);   // (64, 32, 8) — token tiles past count early-exit
  dim3 g2(CAP / BT, D / BC, E);   // (64, 16, 8)

  if (use_tdm) {
    const int wgrid = (int)(wElems / 1024);
    moe_cvt_bf16<<<wgrid, 256, 0, stream>>>(wg, wgb);
    moe_cvt_bf16<<<wgrid, 256, 0, stream>>>(wu, wub);
    moe_cvt_bf16<<<wgrid, 256, 0, stream>>>(wd, wdb);
    moe_gemm_gu_t<true><<<g1, 256, 0, stream>>>(xb, nullptr, nullptr, wgb, wub,
                                                counts, lists, h);
    moe_gemm_down_t<true><<<g2, 256, 0, stream>>>(h, nullptr, wdb,
                                                  counts, lists, wlists, contrib);
  } else {
    moe_gemm_gu_t<false><<<g1, 256, 0, stream>>>(xb, wg, wu, nullptr, nullptr,
                                                 counts, lists, h);
    moe_gemm_down_t<false><<<g2, 256, 0, stream>>>(h, wd, nullptr,
                                                   counts, lists, wlists, contrib);
  }

  moe_reduce<<<(N * D) / 1024, 256, 0, stream>>>(contrib, out);
}
